// GIN_75127567942077
// MI455X (gfx1250) — compile-verified, matrix operand order checked by reading, not run on hardware
//
#include <hip/hip_runtime.h>
#include <hip/hip_bf16.h>

// ---------------------------------------------------------------------------
// GIN on MI455X (gfx1250, wave32, WMMA).
// Memory-bound on the edge scatter (≈5 GB ≈ 215us @ 23.3TB/s; the agg buffer
// [N,256] f32 = 100MB fits in the 192MB L2, so f32 atomics resolve on-chip).
// GEMMs (≈48 GFLOP) use v_wmma_f32_16x16x32_f16 with fp32 accumulation.
// Packed weights are staged once per block into LDS (<=128KB of the 320KB
// WGP LDS) so the 8 waves share one global fetch of B; the kt loop is
// software-pipelined so ds_load latency overlaps the WMMA issue.
// ---------------------------------------------------------------------------

typedef __attribute__((ext_vector_type(16))) _Float16 v16h;
typedef __attribute__((ext_vector_type(8)))  float    v8f;

#define DDIM 128
#define HDIM 256
#define CDIM 40

// ---------------------------------------------------------------------------
__global__ void gin_zero_f32(float* __restrict__ p, long n) {
  long i = (long)blockIdx.x * blockDim.x + threadIdx.x;
  long stride = (long)gridDim.x * blockDim.x;
  for (; i < n; i += stride) p[i] = 0.0f;
}

// agg[dst[e]] += h[src[e]] ; one thread per (edge, 4-float chunk)
__global__ void gin_scatter_add(const float* __restrict__ h,
                                const int* __restrict__ src,
                                const int* __restrict__ dst,
                                float* __restrict__ agg,
                                int chunkLog, long total) {
  long t = (long)blockIdx.x * blockDim.x + threadIdx.x;
  if (t >= total) return;
  long e = t >> chunkLog;
  int  c = (int)(t & ((1l << chunkLog) - 1));
  int  Dm = 4 << chunkLog;
  int  s = src[e];
  int  d = dst[e];
  float4 v = *((const float4*)(h + (size_t)s * Dm) + c);
  float* out = agg + (size_t)d * Dm + (size_t)c * 4;
  unsafeAtomicAdd(out + 0, v.x);   // -> global_atomic_add_f32, resolves in L2
  unsafeAtomicAdd(out + 1, v.y);
  unsafeAtomicAdd(out + 2, v.z);
  unsafeAtomicAdd(out + 3, v.w);
}

// ---------------------------------------------------------------------------
// Pack a f32 row-major weight W[K, Nout] into per-lane v16h B-fragments for
// V_WMMA_F32_16X16X32_F16.  16-bit fragment K-mapping (ISA 7.12.2): lane
// half (lane<16 vs >=16) selects kbase 0/8; elements 0..7 -> K=kbase+j,
// elements 8..15 -> K=16+kbase+(j-8).  Column n = lane&15 within an N-tile.
// Layout: Bp[(((kt*nTiles)+nt)*32 + lane)*16 + j]
__global__ void gin_pack_w(const float* __restrict__ W, _Float16* __restrict__ Bp,
                           int K, int Nout, int nTiles, int total) {
  int idx = blockIdx.x * blockDim.x + threadIdx.x;
  if (idx >= total) return;
  int j    = idx & 15;
  int lane = (idx >> 4) & 31;
  int t3   = idx >> 9;
  int nt   = t3 % nTiles;
  int kt   = t3 / nTiles;
  int kbase = (lane & 16) ? 8 : 0;
  int kk = kt * 32 + ((j < 8) ? (kbase + j) : (16 + kbase + (j - 8)));
  int n  = nt * 16 + (lane & 15);
  float v = (n < Nout && kk < K) ? W[(size_t)kk * Nout + n] : 0.0f;
  Bp[idx] = (_Float16)v;
}

// ---------------------------------------------------------------------------
// Fused GEMM:  Out = act( ((1+eps)*X + AGG) @ W  [+ bias] )
// 256 threads = 8 waves; each wave computes rows [rowBase, rowBase+16).
// Whole packed weight staged into LDS once per block; all KT A-fragments in
// registers; N swept tile-by-tile with the kt loop software-pipelined.
// NFULL: nTiles*16 == Nout (no column guard needed).
template <int KT, bool HAS_AGG, bool HAS_BIAS, bool RELU, bool NFULL>
__global__ void __launch_bounds__(256)
gin_wmma_gemm(const float* __restrict__ X, const float* __restrict__ AGG,
              const float* __restrict__ epsPtr,
              const _Float16* __restrict__ Bp,
              const float* __restrict__ bias,
              float* __restrict__ Out,
              int M, int Nout, int nTiles) {
  constexpr int K = KT * 32;
  extern __shared__ char gin_smem[];

  const int lane    = threadIdx.x & 31;
  const int wave    = threadIdx.x >> 5;
  const int rowBase = blockIdx.x * 128 + wave * 16;
  const int m       = rowBase + (lane & 15);
  const int mc      = (m < M) ? m : (M - 1);        // clamp loads (stores guarded)
  const int kbase   = (lane & 16) ? 8 : 0;
  const float scale = HAS_AGG ? (1.0f + epsPtr[0]) : 1.0f;

  // ---- stage packed B into LDS (uint4 = 16B per copy) ----
  {
    const int cnt = KT * nTiles * 64;               // #uint4 in packed weight
    const uint4* gb = (const uint4*)Bp;
    uint4* sb = (uint4*)gin_smem;
    for (int i = threadIdx.x; i < cnt; i += 256) sb[i] = gb[i];
  }

  // ---- build A fragments (fp32 -> f16, fused (1+eps)*h + agg) ----
  v16h a[KT];
  const float* xrow = X + (size_t)mc * K;
  const float* arow = HAS_AGG ? AGG + (size_t)mc * K : nullptr;
#pragma unroll
  for (int kt = 0; kt < KT; ++kt) {
    int k0 = kt * 32 + kbase;
#pragma unroll
    for (int j = 0; j < 8; ++j) {
      if (HAS_AGG) {
        a[kt][j]     = (_Float16)(scale * xrow[k0 + j]      + arow[k0 + j]);
        a[kt][8 + j] = (_Float16)(scale * xrow[k0 + 16 + j] + arow[k0 + 16 + j]);
      } else {
        a[kt][j]     = (_Float16)xrow[k0 + j];
        a[kt][8 + j] = (_Float16)xrow[k0 + 16 + j];
      }
    }
  }

  __syncthreads();   // LDS weight visible to all waves

  // C/D layout: N = lane&15 ; row = r + 8*(lane>=16), r = accumulator VGPR
  const int  n_sub    = lane & 15;
  const int  mrow0    = rowBase + ((lane & 16) ? 8 : 0);
  const bool fullTile = (rowBase + 16 <= M);        // wave-uniform
  const v16h* sB = (const v16h*)gin_smem;

  for (int nt = 0; nt < nTiles; ++nt) {
    v8f c = {};
    // software-pipelined kt loop: prefetch fragment kt+1 before WMMA on kt,
    // so the wait only needs to drain the older pair of ds_loads.
    v16h bcur = sB[(0 * nTiles + nt) * 32 + lane];
#pragma unroll
    for (int kt = 0; kt < KT; ++kt) {
      v16h bnext;
      if (kt + 1 < KT) bnext = sB[((kt + 1) * nTiles + nt) * 32 + lane];
      c = __builtin_amdgcn_wmma_f32_16x16x32_f16(
          /*neg_a=*/false, a[kt], /*neg_b=*/false, bcur,
          /*c_mod=*/(short)0, c, /*reuse_a=*/false, /*reuse_b=*/false);
      if (kt + 1 < KT) bcur = bnext;
    }
    int n = nt * 16 + n_sub;
    if (NFULL || n < Nout) {
      float bv = HAS_BIAS ? bias[n] : 0.0f;
      float* orow = Out + (size_t)mrow0 * Nout + n;
      if (fullTile) {
#pragma unroll
        for (int r = 0; r < 8; ++r) {
          float v = c[r] + bv;
          if (RELU) v = fmaxf(v, 0.0f);
          orow[(size_t)r * Nout] = v;
        }
      } else {
#pragma unroll
        for (int r = 0; r < 8; ++r) {
          if (mrow0 + r < M) {
            float v = c[r] + bv;
            if (RELU) v = fmaxf(v, 0.0f);
            orow[(size_t)r * Nout] = v;
          }
        }
      }
    }
  }
}

// ---------------------------------------------------------------------------
extern "C" void kernel_launch(void* const* d_in, const int* in_sizes, int n_in,
                              void* d_out, int out_size, void* d_ws, size_t ws_size,
                              hipStream_t stream) {
  const float* x    = (const float*)d_in[0];
  const int*   src  = (const int*)d_in[1];
  const int*   dst  = (const int*)d_in[2];
  const float* eps1 = (const float*)d_in[3];
  const float* w1a  = (const float*)d_in[4];
  const float* w1b  = (const float*)d_in[5];
  const float* eps2 = (const float*)d_in[6];
  const float* w2a  = (const float*)d_in[7];
  const float* w2b  = (const float*)d_in[8];
  const float* fcw  = (const float*)d_in[9];
  const float* fcb  = (const float*)d_in[10];

  const int  N = in_sizes[0] / DDIM;
  const long E = in_sizes[1];

  // workspace layout
  char*  ws  = (char*)d_ws;
  size_t off = 0;
  auto alloc = [&](size_t bytes) -> char* {
    char* p = ws + off;
    off += (bytes + 255) & ~(size_t)255;
    return p;
  };
  float* agg  = (float*)alloc((size_t)N * HDIM * sizeof(float));
  float* tbuf = (float*)alloc((size_t)N * HDIM * sizeof(float));
  float* hbuf = (float*)alloc((size_t)N * HDIM * sizeof(float));

  const int nT_H  = HDIM / 16;                // 16
  const int nT_C  = (CDIM + 15) / 16;         // 3 (zero-padded cols 40..47)
  const int sz1a  = (DDIM / 32) * nT_H * 512; // 32768 halves
  const int szHH  = (HDIM / 32) * nT_H * 512; // 65536 halves
  const int szFC  = (HDIM / 32) * nT_C * 512; // 12288 halves
  _Float16* p1a = (_Float16*)alloc((size_t)sz1a * 2);
  _Float16* p1b = (_Float16*)alloc((size_t)szHH * 2);
  _Float16* p2a = (_Float16*)alloc((size_t)szHH * 2);
  _Float16* p2b = (_Float16*)alloc((size_t)szHH * 2);
  _Float16* pfc = (_Float16*)alloc((size_t)szFC * 2);

  // pack weights into WMMA B-fragment layout
  gin_pack_w<<<(sz1a + 255) / 256, 256, 0, stream>>>(w1a, p1a, DDIM, HDIM, nT_H, sz1a);
  gin_pack_w<<<(szHH + 255) / 256, 256, 0, stream>>>(w1b, p1b, HDIM, HDIM, nT_H, szHH);
  gin_pack_w<<<(szHH + 255) / 256, 256, 0, stream>>>(w2a, p2a, HDIM, HDIM, nT_H, szHH);
  gin_pack_w<<<(szHH + 255) / 256, 256, 0, stream>>>(w2b, p2b, HDIM, HDIM, nT_H, szHH);
  gin_pack_w<<<(szFC + 255) / 256, 256, 0, stream>>>(fcw, pfc, HDIM, CDIM, nT_C, szFC);

  const int mBlocks = (N + 127) / 128;
  const size_t sh1a = (size_t)sz1a * 2;   //  64 KB LDS
  const size_t shHH = (size_t)szHH * 2;   // 128 KB LDS
  const size_t shFC = (size_t)szFC * 2;   //  24 KB LDS

  // ---- layer 1 ----
  long z1 = (long)N * DDIM;
  gin_zero_f32<<<2048, 256, 0, stream>>>(agg, z1);
  long tot1 = E * (DDIM / 4);
  gin_scatter_add<<<(unsigned)((tot1 + 255) / 256), 256, 0, stream>>>(x, src, dst, agg, 5, tot1);
  // t = relu(((1+eps1)x + agg) @ w1a)
  gin_wmma_gemm<4, true, false, true, true>
      <<<mBlocks, 256, sh1a, stream>>>(x, agg, eps1, p1a, nullptr, tbuf, N, HDIM, nT_H);
  // h1 = relu(t @ w1b)   (inner matmul + outer layer relu)
  gin_wmma_gemm<8, false, false, true, true>
      <<<mBlocks, 256, shHH, stream>>>(tbuf, nullptr, nullptr, p1b, nullptr, hbuf, N, HDIM, nT_H);

  // ---- layer 2 ----
  long z2 = (long)N * HDIM;
  gin_zero_f32<<<2048, 256, 0, stream>>>(agg, z2);
  long tot2 = E * (HDIM / 4);
  gin_scatter_add<<<(unsigned)((tot2 + 255) / 256), 256, 0, stream>>>(hbuf, src, dst, agg, 6, tot2);
  // t = relu(((1+eps2)h1 + agg) @ w2a)
  gin_wmma_gemm<8, true, false, true, true>
      <<<mBlocks, 256, shHH, stream>>>(hbuf, agg, eps2, p2a, nullptr, tbuf, N, HDIM, nT_H);
  // h2 = relu(t @ w2b)  -> reuse agg buffer (dead after previous GEMM)
  gin_wmma_gemm<8, false, false, true, true>
      <<<mBlocks, 256, shHH, stream>>>(tbuf, nullptr, nullptr, p2b, nullptr, agg, N, HDIM, nT_H);

  // ---- classifier: out = h2 @ fc_w + fc_b ----
  gin_wmma_gemm<8, false, true, false, false>
      <<<mBlocks, 256, shFC, stream>>>(agg, nullptr, nullptr, pfc, fcb, (float*)d_out,
                                       N, CDIM, nT_C);
}
